// RelMultiHeadAttention_12086037971332
// MI455X (gfx1250) — compile-verified, matrix operand order checked by reading, not run on hardware
//
#include <hip/hip_runtime.h>
#include <hip/hip_bf16.h>

#define B_   8
#define N_   512
#define DIM_ 1024
#define H_   16
#define HD_  64
#define REL_ 32
#define BN_  (B_ * N_)

typedef __attribute__((ext_vector_type(16))) __bf16 v16bf;
typedef __attribute__((ext_vector_type(8)))  float  v8f;

__device__ __forceinline__ unsigned short f2bf(float f) {
    unsigned u = __builtin_bit_cast(unsigned, f);
    u += 0x7fffu + ((u >> 16) & 1u);           // round-to-nearest-even
    return (unsigned short)(u >> 16);
}
__device__ __forceinline__ __bf16 us2bf(unsigned short u) {
    return __builtin_bit_cast(__bf16, u);
}
// K-index of fragment element e for this lane-half (ISA 7.12.2 16-bit A layout)
__device__ __forceinline__ int kmap(int e, int half) {
    return e + 8 * half + ((e >= 8) ? 8 : 0);
}

// 32 contiguous K bf16 at base_row -> one 16-bit WMMA operand fragment
__device__ __forceinline__ v16bf load_frag(const unsigned short* base_row, int half) {
    v16bf f;
    const unsigned short* p0 = base_row + 8 * half;       // e = 0..7  -> K = 8h..8h+7
    const unsigned short* p1 = base_row + 16 + 8 * half;  // e = 8..15 -> K = 16+8h..
#pragma unroll
    for (int e = 0; e < 8; ++e) { f[e] = us2bf(p0[e]); f[e + 8] = us2bf(p1[e]); }
    return f;
}

// K strided by `stride` elements (for loading V columns out of an LDS tile)
__device__ __forceinline__ v16bf load_frag_strided(const unsigned short* base, size_t stride, int half) {
    v16bf f;
#pragma unroll
    for (int e = 0; e < 16; ++e) f[e] = us2bf(base[(size_t)kmap(e, half) * stride]);
    return f;
}

__device__ __forceinline__ v8f wmma_bf16(v16bf a, v16bf b, v8f c) {
    return __builtin_amdgcn_wmma_f32_16x16x32_bf16(false, a, false, b, (short)0, c, false, false);
}

// ---- CDNA5 async global->LDS copy (ASYNCcnt path, CDNA5 ISA)
__device__ __forceinline__ unsigned lds_off(const void* p) {
    return (unsigned)(uintptr_t)p;   // low 32 bits of generic LDS address = LDS offset
}
__device__ __forceinline__ void async_load_b128(unsigned ldsaddr, const void* gaddr) {
    asm volatile("global_load_async_to_lds_b128 %0, %1, off"
                 :: "v"(ldsaddr), "v"(gaddr) : "memory");
}
__device__ __forceinline__ void wait_async_all() {
    asm volatile("s_wait_asynccnt 0x0" ::: "memory");
}

// ---------------------------------------------------------------- conversion
__global__ void conv_f32_bf16(const float* __restrict__ src, unsigned short* __restrict__ dst, size_t n) {
    size_t i = (size_t)blockIdx.x * blockDim.x + threadIdx.x;
    size_t stride = (size_t)gridDim.x * blockDim.x;
    for (; i < n; i += stride) dst[i] = f2bf(src[i]);
}

// ------------------------------------------------- NT GEMM: C = A * B^T + bias
// A: M x K bf16 row-major, Bm: N x K bf16 row-major. 4 waves/block.
// 64x64 block tile; A/B k-slices staged to LDS with async-to-LDS loads so all
// 4 waves share each B tile; WMMA operands read from LDS.
template <bool OUT_BF16>
__global__ void __launch_bounds__(128) gemm_nt(const unsigned short* __restrict__ A,
                                               const unsigned short* __restrict__ Bm,
                                               const float* __restrict__ bias,
                                               void* __restrict__ Cout,
                                               int M, int N, int K) {
    __shared__ unsigned short sA[64][32];   // 4 KB
    __shared__ unsigned short sB[64][32];   // 4 KB
    const int tid  = threadIdx.x;
    const int lane = tid & 31;
    const int wave = tid >> 5;
    const int half = lane >> 4;
    const int row  = lane & 15;
    const int m0b = blockIdx.x * 64;
    const int n0  = blockIdx.y * 64;

    v8f acc[4];
#pragma unroll
    for (int t = 0; t < 4; ++t)
#pragma unroll
        for (int j = 0; j < 8; ++j) acc[t][j] = 0.f;

#pragma unroll 1
    for (int k0 = 0; k0 < K; k0 += 32) {
        __syncthreads();   // previous tile fully consumed
        // each tile = 64 rows x 64 B = 256 x 16 B chunks; 128 threads x 2 chunks
#pragma unroll
        for (int i = 0; i < 2; ++i) {
            int c  = tid + i * 128;
            int r  = c >> 2;
            int cc = c & 3;
            async_load_b128(lds_off(&sA[r][cc * 8]),
                            A + (size_t)(m0b + r) * K + k0 + cc * 8);
            async_load_b128(lds_off(&sB[r][cc * 8]),
                            Bm + (size_t)(n0 + r) * K + k0 + cc * 8);
        }
        wait_async_all();
        __syncthreads();   // all waves' async copies have landed
        v16bf a = load_frag(&sA[wave * 16 + row][0], half);
#pragma unroll
        for (int t = 0; t < 4; ++t)
            acc[t] = wmma_bf16(a, load_frag(&sB[t * 16 + row][0], half), acc[t]);
    }
#pragma unroll
    for (int t = 0; t < 4; ++t) {
        int col = n0 + t * 16 + row;        // C column = lane&15
        float bv = bias[col];
#pragma unroll
        for (int j = 0; j < 8; ++j) {       // C row = j + 8*half
            int m = m0b + wave * 16 + j + 8 * half;
            float v = acc[t][j] + bv;
            if (OUT_BF16) ((unsigned short*)Cout)[(size_t)m * N + col] = f2bf(v);
            else          ((float*)Cout)[(size_t)m * N + col] = v;
        }
    }
}

// ------------------------- rel projection: rel_out[b,h,n,m] = rel_bias[b,n,m,:]·W_rel[h,:] + b_rel[h]
// One wave per (b, n, half-of-m): M=16 m-values, N=16 heads, K=32; 16 WMMAs per wave.
__global__ void __launch_bounds__(32) rel_proj(const float* __restrict__ rel_bias,
                                               const float* __restrict__ W_rel,
                                               const float* __restrict__ b_rel,
                                               float* __restrict__ rel_out) {
    const int lane = threadIdx.x & 31;
    const int half = lane >> 4;
    const int row  = lane & 15;
    int id  = blockIdx.x;
    int seg = id & 1;
    int n   = (id >> 1) % N_;
    int b   = (id >> 1) / N_;

    // B fragment: W_rel row per lane (head = lane&15), f32 -> bf16 in-register
    v16bf wb;
    const float* wr = W_rel + row * REL_;
#pragma unroll
    for (int e = 0; e < 16; ++e) wb[e] = us2bf(f2bf(wr[kmap(e, half)]));
    float brel = b_rel[row];

#pragma unroll 1
    for (int mt = 0; mt < 16; ++mt) {
        int m0 = seg * 256 + mt * 16;
        const float* ar = rel_bias + (((size_t)b * N_ + n) * N_ + (m0 + row)) * REL_;
        v16bf a;
#pragma unroll
        for (int e = 0; e < 16; ++e) a[e] = us2bf(f2bf(ar[kmap(e, half)]));
        v8f acc;
#pragma unroll
        for (int j = 0; j < 8; ++j) acc[j] = 0.f;
        acc = wmma_bf16(a, wb, acc);
#pragma unroll
        for (int j = 0; j < 8; ++j) {
            int m = m0 + j + 8 * half;
            rel_out[(((size_t)b * H_ + row) * N_ + n) * N_ + m] = acc[j] + brel;
        }
    }
}

// --------------------------------------------------------------- attention
// One wave per (b, h, 16-row tile): scores = q k^T (WMMA) * scale + rel,
// vectorized f32 softmax in LDS, then P @ V with V k-chunks async-staged to
// LDS (double buffered) so the column accesses are DS reads, not strided VMEM.
__global__ void __launch_bounds__(32) attn_kernel(const unsigned short* __restrict__ qkv,
                                                  const float* __restrict__ rel,
                                                  unsigned short* __restrict__ o) {
    __shared__ float          sS[16][N_];       // 32 KB scores
    __shared__ unsigned short sP[16][N_];       // 16 KB probs (bf16)
    __shared__ unsigned short sV[2][32][HD_];   //  8 KB V chunk, double buffered

    const int lane = threadIdx.x & 31;
    const int half = lane >> 4;
    const int row  = lane & 15;
    int id = blockIdx.x;
    int nt = id & 31;            // N_/16 = 32 tiles
    int h  = (id >> 5) & 15;
    int b  = id >> 9;
    int n0 = nt * 16;
    const size_t ldq = 3 * DIM_;

    // Q fragments (K = head_dim = 64 -> two K=32 fragments), reused for all m-tiles
    const unsigned short* qrow = qkv + (size_t)(b * N_ + n0 + row) * ldq + h * HD_;
    v16bf a0 = load_frag(qrow, half);
    v16bf a1 = load_frag(qrow + 32, half);

    const float scale = 0.125f;  // hd^-0.5
    const float* relbase = rel + ((size_t)(b * H_ + h) * N_ + n0) * N_;

#pragma unroll 1
    for (int mt = 0; mt < 32; ++mt) {
        int m0 = mt * 16;
        const unsigned short* krow = qkv + (size_t)(b * N_ + m0 + row) * ldq + DIM_ + h * HD_;
        v8f acc;
#pragma unroll
        for (int j = 0; j < 8; ++j) acc[j] = 0.f;
        acc = wmma_bf16(a0, load_frag(krow, half), acc);
        acc = wmma_bf16(a1, load_frag(krow + 32, half), acc);
#pragma unroll
        for (int j = 0; j < 8; ++j) {
            int rr = j + 8 * half;
            sS[rr][m0 + row] = acc[j] * scale + relbase[(size_t)rr * N_ + m0 + row];
        }
    }
    __syncthreads();

    // softmax: two lanes per row (lane & lane+16), 256 columns each, float4-wide
    {
        float4* rowp = reinterpret_cast<float4*>(&sS[row][half * 256]);
        float mx = -3.4e38f;
#pragma unroll 1
        for (int c = 0; c < 64; ++c) {
            float4 v = rowp[c];
            mx = fmaxf(fmaxf(fmaxf(mx, v.x), v.y), fmaxf(v.z, v.w));
        }
        mx = fmaxf(mx, __shfl_xor(mx, 16, 32));
        float sum = 0.f;
#pragma unroll 1
        for (int c = 0; c < 64; ++c) {
            float4 v = rowp[c];
            v.x = __expf(v.x - mx); v.y = __expf(v.y - mx);
            v.z = __expf(v.z - mx); v.w = __expf(v.w - mx);
            rowp[c] = v;
            sum += (v.x + v.y) + (v.z + v.w);
        }
        sum += __shfl_xor(sum, 16, 32);
        float inv = 1.f / sum;
        unsigned short* prow = &sP[row][half * 256];
#pragma unroll 1
        for (int c = 0; c < 64; ++c) {
            float4 v = rowp[c];
            uint2 pk;
            pk.x = (unsigned)f2bf(v.x * inv) | ((unsigned)f2bf(v.y * inv) << 16);
            pk.y = (unsigned)f2bf(v.z * inv) | ((unsigned)f2bf(v.w * inv) << 16);
            *reinterpret_cast<uint2*>(&prow[4 * c]) = pk;
        }
    }
    __syncthreads();

    // out = P @ V : K = 512 (16 chunks of 32), four 16-wide d tiles.
    v8f acc[4];
#pragma unroll
    for (int t = 0; t < 4; ++t)
#pragma unroll
        for (int j = 0; j < 8; ++j) acc[t][j] = 0.f;
#pragma unroll 1
    for (int kt = 0; kt < 16; ++kt) {
        int buf = kt & 1;
        // stage V rows kt*32..+31, cols h*64..+63 : 32 x 128 B = 256 chunks / 32 lanes
#pragma unroll
        for (int i = 0; i < 8; ++i) {
            int c  = i * 32 + lane;
            int r  = c >> 3;
            int cc = c & 7;
            async_load_b128(lds_off(&sV[buf][r][cc * 8]),
                            qkv + (size_t)(b * N_ + kt * 32 + r) * ldq + 2 * DIM_ + h * HD_ + cc * 8);
        }
        wait_async_all();   // single wave per block: no barrier needed
        v16bf a = load_frag(&sP[row][kt * 32], half);
#pragma unroll
        for (int t = 0; t < 4; ++t) {
            const unsigned short* vb = &sV[buf][0][t * 16 + row];
            acc[t] = wmma_bf16(a, load_frag_strided(vb, HD_, half), acc[t]);
        }
    }
#pragma unroll
    for (int t = 0; t < 4; ++t)
#pragma unroll
        for (int j = 0; j < 8; ++j) {
            int n = n0 + j + 8 * half;
            o[(size_t)(b * N_ + n) * DIM_ + h * HD_ + t * 16 + row] = f2bf(acc[t][j]);
        }
}

// ---------------------------------------------------------------- launcher
extern "C" void kernel_launch(void* const* d_in, const int* in_sizes, int n_in,
                              void* d_out, int out_size, void* d_ws, size_t ws_size,
                              hipStream_t stream) {
    (void)in_sizes; (void)n_in; (void)out_size; (void)ws_size;
    const float* x        = (const float*)d_in[0];
    const float* rel_bias = (const float*)d_in[1];
    const float* W_qkv    = (const float*)d_in[2];
    const float* b_qkv    = (const float*)d_in[3];
    const float* W_rel    = (const float*)d_in[4];
    const float* b_rel    = (const float*)d_in[5];
    const float* W_out    = (const float*)d_in[6];
    const float* b_out    = (const float*)d_in[7];
    float* out = (float*)d_out;

    char* ws = (char*)d_ws;
    size_t off = 0;
    unsigned short* x_bf    = (unsigned short*)(ws + off); off += (size_t)BN_ * DIM_ * 2;       // 8 MB
    unsigned short* wqkv_bf = (unsigned short*)(ws + off); off += (size_t)3 * DIM_ * DIM_ * 2;  // 6 MB
    unsigned short* wout_bf = (unsigned short*)(ws + off); off += (size_t)DIM_ * DIM_ * 2;      // 2 MB
    unsigned short* qkv_bf  = (unsigned short*)(ws + off); off += (size_t)BN_ * 3 * DIM_ * 2;   // 24 MB
    float*          rel_f   = (float*)(ws + off);          off += (size_t)B_ * H_ * N_ * N_ * 4;// 128 MB
    unsigned short* o_bf    = (unsigned short*)(ws + off); off += (size_t)BN_ * DIM_ * 2;       // 8 MB

    // 1) f32 -> bf16 conversions
    conv_f32_bf16<<<1024, 256, 0, stream>>>(x, x_bf, (size_t)BN_ * DIM_);
    conv_f32_bf16<<<1024, 256, 0, stream>>>(W_qkv, wqkv_bf, (size_t)3 * DIM_ * DIM_);
    conv_f32_bf16<<<256, 256, 0, stream>>>(W_out, wout_bf, (size_t)DIM_ * DIM_);

    // 2) qkv = x @ W_qkv^T + b_qkv  (bf16 out)
    {
        dim3 g(BN_ / 64, (3 * DIM_) / 64);
        gemm_nt<true><<<g, 128, 0, stream>>>(x_bf, wqkv_bf, b_qkv, (void*)qkv_bf,
                                             BN_, 3 * DIM_, DIM_);
    }

    // 3) rel[b,h,n,m] (f32, read rel_bias exactly once)
    rel_proj<<<B_ * N_ * 2, 32, 0, stream>>>(rel_bias, W_rel, b_rel, rel_f);

    // 4) attention -> o (bf16, layout (B*N, DIM))
    attn_kernel<<<B_ * H_ * (N_ / 16), 32, 0, stream>>>(qkv_bf, rel_f, o_bf);

    // 5) out = o @ W_out^T + b_out (f32 to d_out)
    {
        dim3 g(BN_ / 64, DIM_ / 64);
        gemm_nt<false><<<g, 128, 0, stream>>>(o_bf, wout_bf, b_out, (void*)out,
                                              BN_, DIM_, DIM_);
    }
}
